// ViT_33801392619588
// MI455X (gfx1250) — matrix operand order repaired and checked by reading, not compiled, not run on hardware
//
#include <hip/hip_runtime.h>
#include <hip/hip_bf16.h>
#include <math.h>

// ---------------------------------------------------------------------------
// ViT-Base forward for MI455X (gfx1250, wave32, WMMA bf16, fp32 accumulate).
// Guard-free vectorized LDS staging; ds_load_b128 fragment gathers.
// ---------------------------------------------------------------------------

typedef __attribute__((ext_vector_type(16))) __bf16        v16bf;
typedef __attribute__((ext_vector_type(8)))  float         v8f;
typedef __attribute__((ext_vector_type(4)))  unsigned int  u32x4;

#define HD     768
#define NHD    8
#define DH     96
#define SEQ    197
#define SPAD   224              // 7*32: every attention dim is a K/M multiple
#define BATCH  32
#define LAYERS 12
#define EXPH   3072
#define NPATCH 196
#define BS     (BATCH * SEQ)    // 6304 = 197*32
#define MPATCH (BATCH * NPATCH) // 6272 = 196*32
#define BN     (BATCH * NHD)    // 256 attention batches

__device__ __forceinline__ unsigned short f2bf(float f) {
  unsigned int u = __float_as_uint(f);
  u += 0x7fffu + ((u >> 16) & 1u);     // round-to-nearest-even
  return (unsigned short)(u >> 16);
}

// ---------------------------------------------------------------------------
// WMMA GEMM:  C[M,N] (+)= A[M,K](bf16) * B(bf16) + bias
//  - A row-major [M,K] (lda), B stored [N,K] row-major (ldb)  ("B-transposed")
//  - Caller invariants: M % 32 == 0 (grid.y = M/32 exact), K % 32 == 0,
//    B buffer has >= gridDim.x*128 rows allocated (padded; extra rows are
//    never stored: C-store guards gn < N).
//  - Block 256 threads = 8 waves; block tile 32(M) x 128(N); wave tile 16x32
//    (two v_wmma per K-step). Batched via blockIdx.z (element strides).
// ---------------------------------------------------------------------------
template <bool ACC>
__global__ __launch_bounds__(256) void gemm_bf16(
    const unsigned short* __restrict__ A, int lda, long sA,
    const unsigned short* __restrict__ Bm, int ldb, long sB,
    const float* __restrict__ bias,
    float* __restrict__ C, int ldc, long sC,
    int M, int N, int K)
{
  __shared__ __align__(16) unsigned short As[32][40];    // [m][k], 80B rows
  __shared__ __align__(16) unsigned short BsT[128][40];  // [n][k], 80B rows

  const int tid  = threadIdx.x;
  const int lane = tid & 31;
  const int wave = tid >> 5;
  const int mw   = wave >> 2;            // 0..1
  const int nw   = wave & 3;             // 0..3 (32 cols each)
  const int m0   = blockIdx.y * 32;
  const int n0   = blockIdx.x * 128;

  A  += (long)blockIdx.z * sA;
  Bm += (long)blockIdx.z * sB;
  C  += (long)blockIdx.z * sC;

  // staging maps (all unguarded, fully in-bounds by caller invariants)
  const int ar = tid >> 3, ac = (tid & 7) * 4;           // A: b64 per thread
  const int bn = tid >> 1, bk = (tid & 1) * 16;          // B: 2x b128 per thread
  const long aRowBase = (long)(m0 + ar) * lda;
  const long bRowBase = (long)(n0 + bn) * ldb;

  v8f acc0 = {}, acc1 = {};

  for (int k0 = 0; k0 < K; k0 += 32) {
    *(unsigned long long*)&As[ar][ac] =
        *(const unsigned long long*)&A[aRowBase + k0 + ac];
    *(u32x4*)&BsT[bn][bk] = *(const u32x4*)&Bm[bRowBase + k0 + bk];
    *(u32x4*)&BsT[bn][bk + 8] = *(const u32x4*)&Bm[bRowBase + k0 + bk + 8];
    __syncthreads();

    if (k0 + 32 < K)  // global_prefetch_b8 of next A slab
      __builtin_prefetch(&A[aRowBase + k0 + 32 + ac], 0, 1);

    // A 16x32 frag: lane<16 -> M=lane, K 0-15; lane>=16 -> K 16-31.
    union { v16bf v; u32x4 q[2]; } fa, fb0, fb1;
    const int mrow = mw * 16 + (lane & 15);
    const int ka   = (lane >> 4) * 16;
    fa.q[0] = *(const u32x4*)&As[mrow][ka];
    fa.q[1] = *(const u32x4*)&As[mrow][ka + 8];
    // B 32x16 frag: lane<16 -> N=lane, K 0-15; lane>=16 -> K 16-31.
    const int nc = nw * 32 + (lane & 15);
    const int kb = (lane >> 4) * 16;
    fb0.q[0] = *(const u32x4*)&BsT[nc][kb];
    fb0.q[1] = *(const u32x4*)&BsT[nc][kb + 8];
    fb1.q[0] = *(const u32x4*)&BsT[nc + 16][kb];
    fb1.q[1] = *(const u32x4*)&BsT[nc + 16][kb + 8];

    acc0 = __builtin_amdgcn_wmma_f32_16x16x32_bf16(
        false, fa.v, false, fb0.v, (short)0, acc0, false, false);
    acc1 = __builtin_amdgcn_wmma_f32_16x16x32_bf16(
        false, fa.v, false, fb1.v, (short)0, acc1, false, false);
    __syncthreads();
  }

  // C/D layout: VGPR r -> M = r + 8*(lane>=16), N = lane&15.
  const int gm0 = m0 + mw * 16 + (lane >> 4) * 8;
  const int gn0 = n0 + nw * 32 + (lane & 15);
  const int gn1 = gn0 + 16;
  const float b0 = (bias && gn0 < N) ? bias[gn0] : 0.0f;
  const float b1 = (bias && gn1 < N) ? bias[gn1] : 0.0f;
#pragma unroll
  for (int r = 0; r < 8; ++r) {
    const int gm = gm0 + r;
    if (gn0 < N) {
      const long i = (long)gm * ldc + gn0;
      float v = acc0[r] + b0;
      if (ACC) v += C[i];
      C[i] = v;
    }
    if (gn1 < N) {
      const long i = (long)gm * ldc + gn1;
      float v = acc1[r] + b1;
      if (ACC) v += C[i];
      C[i] = v;
    }
  }
}

// ---------------------------------------------------------------------------
// Support kernels
// ---------------------------------------------------------------------------
__global__ void f32_to_bf16_k(const float* __restrict__ x,
                              unsigned short* __restrict__ y, long n) {
  for (long i = blockIdx.x * (long)blockDim.x + threadIdx.x; i < n;
       i += (long)gridDim.x * blockDim.x)
    y[i] = f2bf(x[i]);
}

// src f32 [K,N] row-major -> dst bf16 [Npad,K] row-major ([N,K] + zero pad)
__global__ void transpose_bf16_k(const float* __restrict__ src,
                                 unsigned short* __restrict__ dst,
                                 int K, int N, int Npad) {
  const long total = (long)Npad * K;
  for (long idx = blockIdx.x * (long)blockDim.x + threadIdx.x; idx < total;
       idx += (long)gridDim.x * blockDim.x) {
    const int n = (int)(idx / K), k = (int)(idx % K);
    dst[idx] = (n < N) ? f2bf(src[(long)k * N + n]) : (unsigned short)0;
  }
}

// x[B,3,224,224] -> Ap[6272,768] bf16 ; row = b*196 + (w*14+h),
// col = c*256 + a*16 + bb   (matches einsum 'ncwahb,ocab->nwho')
__global__ void patchify_k(const float* __restrict__ x,
                           unsigned short* __restrict__ Ap) {
  const long total = (long)MPATCH * HD;
  for (long idx = blockIdx.x * (long)blockDim.x + threadIdx.x; idx < total;
       idx += (long)gridDim.x * blockDim.x) {
    const int m = (int)(idx / HD), k = (int)(idx % HD);
    const int n = m / NPATCH, p = m % NPATCH;
    const int w = p / 14, hcol = p % 14;
    const int c = k >> 8, rem = k & 255, a = rem >> 4, bcol = rem & 15;
    const long src = (((long)n * 3 + c) * 224 + (w * 16 + a)) * 224
                     + (hcol * 16 + bcol);
    Ap[idx] = f2bf(x[src]);
  }
}

__global__ void assemble_h_k(const float* __restrict__ emb,
                             const float* __restrict__ cls,
                             float* __restrict__ h) {
  const long total = (long)BS * HD;
  for (long idx = blockIdx.x * (long)blockDim.x + threadIdx.x; idx < total;
       idx += (long)gridDim.x * blockDim.x) {
    const int m = (int)(idx / HD), hc = (int)(idx % HD);
    const int n = m / SEQ, s = m % SEQ;
    h[idx] = (s == 0) ? cls[hc]
                      : emb[((long)n * NPATCH + (s - 1)) * HD + hc];
  }
}

__global__ __launch_bounds__(256) void layernorm_bf16_k(
    const float* __restrict__ x, const float* __restrict__ g,
    const float* __restrict__ b, unsigned short* __restrict__ y, int H) {
  __shared__ float red[256];
  const long row = blockIdx.x;
  const float* xr = x + row * H;
  float s = 0.f;
  for (int i = threadIdx.x; i < H; i += 256) s += xr[i];
  red[threadIdx.x] = s; __syncthreads();
  for (int o = 128; o > 0; o >>= 1) {
    if (threadIdx.x < o) red[threadIdx.x] += red[threadIdx.x + o];
    __syncthreads();
  }
  const float mu = red[0] / H; __syncthreads();
  float vs = 0.f;
  for (int i = threadIdx.x; i < H; i += 256) {
    const float d = xr[i] - mu; vs += d * d;
  }
  red[threadIdx.x] = vs; __syncthreads();
  for (int o = 128; o > 0; o >>= 1) {
    if (threadIdx.x < o) red[threadIdx.x] += red[threadIdx.x + o];
    __syncthreads();
  }
  const float inv = rsqrtf(red[0] / H + 1e-5f);
  unsigned short* yr = y + row * H;
  for (int i = threadIdx.x; i < H; i += 256)
    yr[i] = f2bf((xr[i] - mu) * inv * g[i] + b[i]);
}

// [B*S,768] f32 -> [BN, R, 96] bf16 rows s<SPAD (zero pad s>=SEQ). R=row cap.
__global__ void split_qk_k(const float* __restrict__ src,
                           unsigned short* __restrict__ dst, int R) {
  const long total = (long)BN * SPAD * DH;
  for (long idx = blockIdx.x * (long)blockDim.x + threadIdx.x; idx < total;
       idx += (long)gridDim.x * blockDim.x) {
    const int dd = (int)(idx % DH);
    long t = idx / DH;
    const int s = (int)(t % SPAD);
    const int bnb = (int)(t / SPAD);
    const int n = bnb >> 3, hh = bnb & 7;
    float v = 0.f;
    if (s < SEQ) v = src[((long)n * SEQ + s) * HD + hh * DH + dd];
    dst[((long)bnb * R + s) * DH + dd] = f2bf(v);
  }
}

// V head-transposed: [BN, 128, SPAD] bf16, dst[bn][dd][s] = v(s,dd)
__global__ void split_v_k(const float* __restrict__ src,
                          unsigned short* __restrict__ dst) {
  const long total = (long)BN * SPAD * DH;
  for (long idx = blockIdx.x * (long)blockDim.x + threadIdx.x; idx < total;
       idx += (long)gridDim.x * blockDim.x) {
    const int dd = (int)(idx % DH);
    long t = idx / DH;
    const int s = (int)(t % SPAD);
    const int bnb = (int)(t / SPAD);
    const int n = bnb >> 3, hh = bnb & 7;
    float v = 0.f;
    if (s < SEQ) v = src[((long)n * SEQ + s) * HD + hh * DH + dd];
    dst[((long)bnb * 128 + dd) * SPAD + s] = f2bf(v);
  }
}

// Faithful quirk: softmax across the 8 HEADS at each (b,q,k), then / sqrt(768)
__global__ void softmax_heads_k(const float* __restrict__ e,
                                unsigned short* __restrict__ att) {
  const long total = (long)BATCH * SPAD * SPAD;
  const long hs = (long)SPAD * SPAD;
  for (long idx = blockIdx.x * (long)blockDim.x + threadIdx.x; idx < total;
       idx += (long)gridDim.x * blockDim.x) {
    const int ki = (int)(idx % SPAD);
    long t = idx / SPAD;
    const int qi = (int)(t % SPAD);
    const int n = (int)(t / SPAD);
    const long base = (((long)n * NHD) * SPAD + qi) * SPAD + ki;
    float mx = -3.0e38f;
    for (int hh = 0; hh < NHD; ++hh) mx = fmaxf(mx, e[base + hh * hs]);
    float ex[NHD], sum = 0.f;
    for (int hh = 0; hh < NHD; ++hh) {
      ex[hh] = expf(e[base + hh * hs] - mx); sum += ex[hh];
    }
    const float inv = 1.0f / (sum * 27.712812921102035f);  // * sqrt(768)
    for (int hh = 0; hh < NHD; ++hh) att[base + hh * hs] = f2bf(ex[hh] * inv);
  }
}

// [BN,224,96] f32 -> [B*S,768] bf16 (heads merged)
__global__ void merge_heads_k(const float* __restrict__ oh,
                              unsigned short* __restrict__ om) {
  const long total = (long)BS * HD;
  for (long idx = blockIdx.x * (long)blockDim.x + threadIdx.x; idx < total;
       idx += (long)gridDim.x * blockDim.x) {
    const int hc = (int)(idx % HD);
    const int m = (int)(idx / HD);
    const int n = m / SEQ, s = m % SEQ;
    const int hh = hc / DH, dd = hc % DH;
    om[idx] = f2bf(oh[(((long)n * NHD + hh) * SPAD + s) * DH + dd]);
  }
}

__global__ void gelu_k(const float* __restrict__ u,
                       unsigned short* __restrict__ ub, long n) {
  for (long i = blockIdx.x * (long)blockDim.x + threadIdx.x; i < n;
       i += (long)gridDim.x * blockDim.x) {
    const float v = u[i];
    ub[i] = f2bf(0.5f * v * (1.0f + erff(v * 0.70710678118654752f)));
  }
}

__global__ void pool_mean_k(const float* __restrict__ h,
                            float* __restrict__ p) {
  const int idx = blockIdx.x * blockDim.x + threadIdx.x;
  if (idx >= BATCH * HD) return;
  const int n = idx / HD, hc = idx % HD;
  const float* base = h + ((long)n * SEQ) * HD + hc;
  float s = 0.f;
  for (int t = 0; t < SEQ; ++t) s += base[(long)t * HD];
  p[idx] = s * (1.0f / SEQ);
}

// ---------------------------------------------------------------------------
// Host driver
// ---------------------------------------------------------------------------
static inline int ew_blocks(long n) {
  long b = (n + 255) / 256;
  return (int)(b > 32768 ? 32768 : b);
}

extern "C" void kernel_launch(void* const* d_in, const int* in_sizes, int n_in,
                              void* d_out, int out_size, void* d_ws,
                              size_t ws_size, hipStream_t stream) {
  (void)in_sizes; (void)n_in; (void)out_size; (void)ws_size;
  const float* x      = (const float*)d_in[0];
  const float* conv_w = (const float*)d_in[1];
  const float* conv_b = (const float*)d_in[2];
  const float* cls    = (const float*)d_in[3];
  const float* ln1_g  = (const float*)d_in[4];
  const float* ln1_b  = (const float*)d_in[5];
  const float* wq = (const float*)d_in[6];  const float* bq = (const float*)d_in[7];
  const float* wk = (const float*)d_in[8];  const float* bk = (const float*)d_in[9];
  const float* wv = (const float*)d_in[10]; const float* bv = (const float*)d_in[11];
  const float* wp = (const float*)d_in[12]; const float* bp = (const float*)d_in[13];
  const float* ln2_g = (const float*)d_in[14]; const float* ln2_b = (const float*)d_in[15];
  const float* w1 = (const float*)d_in[16]; const float* b1 = (const float*)d_in[17];
  const float* w2 = (const float*)d_in[18]; const float* b2 = (const float*)d_in[19];
  const float* hln_g = (const float*)d_in[20]; const float* hln_b = (const float*)d_in[21];
  const float* hw = (const float*)d_in[22]; const float* hb = (const float*)d_in[23];

  char* ws = (char*)d_ws;
  size_t off = 0;
  auto alloc = [&](size_t bytes) -> void* {
    void* p = ws + off;
    off += (bytes + 255) & ~(size_t)255;
    return p;
  };
  const long HH = (long)HD * HD, HE = (long)HD * EXPH;
  // transposed bf16 weights, layout [N(pad), K]
  unsigned short* wqT = (unsigned short*)alloc(2 * LAYERS * HH);
  unsigned short* wkT = (unsigned short*)alloc(2 * LAYERS * HH);
  unsigned short* wvT = (unsigned short*)alloc(2 * LAYERS * HH);
  unsigned short* wpT = (unsigned short*)alloc(2 * LAYERS * HH);
  unsigned short* w1T = (unsigned short*)alloc(2 * LAYERS * HE);   // [3072,768]
  unsigned short* w2T = (unsigned short*)alloc(2 * LAYERS * HE);   // [768,3072]
  unsigned short* cwB = (unsigned short*)alloc(2 * HH);            // already [N,K]
  unsigned short* hwT = (unsigned short*)alloc(2L * 1024 * HD);    // [1024,768]
  unsigned short* Ap  = (unsigned short*)alloc(2L * MPATCH * HD);
  float*          emb = (float*)alloc(4L * MPATCH * HD);
  float*          h   = (float*)alloc(4L * BS * HD);
  unsigned short* y   = (unsigned short*)alloc(2L * BS * HD);
  float*          tmp = (float*)alloc(4L * BS * HD);
  unsigned short* qh  = (unsigned short*)alloc(2L * BN * SPAD * DH);
  unsigned short* khT = (unsigned short*)alloc(2L * BN * 256 * DH);   // padded rows
  unsigned short* vhT = (unsigned short*)alloc(2L * BN * 128 * SPAD); // padded rows
  float*          enr = (float*)alloc(4L * BN * SPAD * SPAD);
  unsigned short* att = (unsigned short*)alloc(2L * BN * SPAD * SPAD);
  float*          oh  = (float*)alloc(4L * BN * SPAD * DH);
  unsigned short* om  = (unsigned short*)alloc(2L * BS * HD);
  float*          u   = (float*)alloc(4L * BS * EXPH);
  unsigned short* ub  = (unsigned short*)alloc(2L * BS * EXPH);
  float*          pooled  = (float*)alloc(4L * BATCH * HD);
  unsigned short* pooledb = (unsigned short*)alloc(2L * BATCH * HD);

  // ---- weights -> bf16, transposed to [N,K] ------------------------------
  for (int l = 0; l < LAYERS; ++l) {
    transpose_bf16_k<<<ew_blocks(HH), 256, 0, stream>>>(wq + l * HH, wqT + l * HH, HD, HD, HD);
    transpose_bf16_k<<<ew_blocks(HH), 256, 0, stream>>>(wk + l * HH, wkT + l * HH, HD, HD, HD);
    transpose_bf16_k<<<ew_blocks(HH), 256, 0, stream>>>(wv + l * HH, wvT + l * HH, HD, HD, HD);
    transpose_bf16_k<<<ew_blocks(HH), 256, 0, stream>>>(wp + l * HH, wpT + l * HH, HD, HD, HD);
    transpose_bf16_k<<<ew_blocks(HE), 256, 0, stream>>>(w1 + l * HE, w1T + l * HE, HD, EXPH, EXPH);
    transpose_bf16_k<<<ew_blocks(HE), 256, 0, stream>>>(w2 + l * HE, w2T + l * HE, EXPH, HD, HD);
  }
  f32_to_bf16_k<<<ew_blocks(HH), 256, 0, stream>>>(conv_w, cwB, HH);  // [o,cab]=[N,K]
  transpose_bf16_k<<<ew_blocks(1024L * HD), 256, 0, stream>>>(hw, hwT, HD, 1000, 1024);

  // ---- patch embedding as GEMM ------------------------------------------
  patchify_k<<<ew_blocks((long)MPATCH * HD), 256, 0, stream>>>(x, Ap);
  gemm_bf16<false><<<dim3(HD / 128, MPATCH / 32, 1), 256, 0, stream>>>(
      Ap, HD, 0, cwB, HD, 0, conv_b, emb, HD, 0, MPATCH, HD, HD);
  assemble_h_k<<<ew_blocks((long)BS * HD), 256, 0, stream>>>(emb, cls, h);

  const dim3 gHH(HD / 128, BS / 32, 1);          // 6304 x 768
  const dim3 gM1(EXPH / 128, BS / 32, 1);        // 6304 x 3072
  const dim3 gE(SPAD / 112 + 1, SPAD / 32, BN);  // 3 x 7 x 256
  const dim3 gAV(1, SPAD / 32, BN);              // 1 x 7 x 256
  const long qstride = (long)SPAD * DH;
  const long kstride = 256L * DH;
  const long vstride = 128L * SPAD;
  const long estride = (long)SPAD * SPAD;
  const long nsplit  = (long)BN * SPAD * DH;

  for (int l = 0; l < LAYERS; ++l) {
    layernorm_bf16_k<<<BS, 256, 0, stream>>>(h, ln1_g + l * HD, ln1_b + l * HD, y, HD);

    gemm_bf16<false><<<gHH, 256, 0, stream>>>(
        y, HD, 0, wqT + l * HH, HD, 0, bq + l * HD, tmp, HD, 0, BS, HD, HD);
    split_qk_k<<<ew_blocks(nsplit), 256, 0, stream>>>(tmp, qh, SPAD);
    gemm_bf16<false><<<gHH, 256, 0, stream>>>(
        y, HD, 0, wkT + l * HH, HD, 0, bk + l * HD, tmp, HD, 0, BS, HD, HD);
    split_qk_k<<<ew_blocks(nsplit), 256, 0, stream>>>(tmp, khT, 256);
    gemm_bf16<false><<<gHH, 256, 0, stream>>>(
        y, HD, 0, wvT + l * HH, HD, 0, bv + l * HD, tmp, HD, 0, BS, HD, HD);
    split_v_k<<<ew_blocks(nsplit), 256, 0, stream>>>(tmp, vhT);

    // energy = q @ k^T  (batched over B*NH): A=[224,96], B=[s,96]=[N,K]
    gemm_bf16<false><<<gE, 256, 0, stream>>>(
        qh, DH, qstride, khT, DH, kstride, nullptr, enr, SPAD, estride,
        SPAD, SPAD, DH);
    softmax_heads_k<<<ew_blocks((long)BATCH * SPAD * SPAD), 256, 0, stream>>>(enr, att);
    // o = att @ v : A=[224,224], B=vhT[dd][s]=[N,K], N=96, K=224
    gemm_bf16<false><<<gAV, 256, 0, stream>>>(
        att, SPAD, estride, vhT, SPAD, vstride, nullptr, oh, DH, qstride,
        SPAD, DH, SPAD);
    merge_heads_k<<<ew_blocks((long)BS * HD), 256, 0, stream>>>(oh, om);

    // h += o @ Wp + Bp
    gemm_bf16<true><<<gHH, 256, 0, stream>>>(
        om, HD, 0, wpT + l * HH, HD, 0, bp + l * HD, h, HD, 0, BS, HD, HD);

    // MLP
    layernorm_bf16_k<<<BS, 256, 0, stream>>>(h, ln2_g + l * HD, ln2_b + l * HD, y, HD);
    gemm_bf16<false><<<gM1, 256, 0, stream>>>(
        y, HD, 0, w1T + l * HE, HD, 0, b1 + (long)l * EXPH, u, EXPH, 0,
        BS, EXPH, HD);
    gelu_k<<<ew_blocks((long)BS * EXPH), 256, 0, stream>>>(u, ub, (long)BS * EXPH);
    gemm_bf16<true><<<gHH, 256, 0, stream>>>(
        ub, EXPH, 0, w2T + l * HE, EXPH, 0, b2 + l * HD, h, HD, 0,
        BS, HD, EXPH);
  }

  // ---- head: mean-pool -> LN -> logits -----------------------------------
  pool_mean_k<<<(BATCH * HD + 255) / 256, 256, 0, stream>>>(h, pooled);
  layernorm_bf16_k<<<BATCH, 256, 0, stream>>>(pooled, hln_g, hln_b, pooledb, HD);
  gemm_bf16<false><<<dim3(1024 / 128, 1, 1), 256, 0, stream>>>(
      pooledb, HD, 0, hwT, HD, 0, hb, (float*)d_out, 1000, 0, BATCH, 1000, HD);
}
